// PrefillDecoderLayer_66709432041901
// MI455X (gfx1250) — matrix operand order repaired and checked by reading, not compiled, not run on hardware
//
#include <hip/hip_runtime.h>
#include <cstddef>
#include <cstdint>

// ---------------------------------------------------------------------------
// Types
// ---------------------------------------------------------------------------
typedef unsigned short u16;
typedef __attribute__((ext_vector_type(8)))  unsigned short u16x8;
typedef __attribute__((ext_vector_type(16))) unsigned short u16x16;
typedef __attribute__((ext_vector_type(16))) __bf16        bf16x16;
typedef __attribute__((ext_vector_type(8)))  float         f32x8;

__device__ __forceinline__ u16 f2bf(float f) {
  unsigned u = __builtin_bit_cast(unsigned, f);
  unsigned r = u + 0x7FFFu + ((u >> 16) & 1u);   // round-to-nearest-even
  return (u16)(r >> 16);
}

__device__ __forceinline__ f32x8 zero8() {
  f32x8 z = {0.f, 0.f, 0.f, 0.f, 0.f, 0.f, 0.f, 0.f};
  return z;
}

// WMMA: D = A(16x32 bf16) * B(32x16 bf16) + C(16x16 f32).  a0/a1, b0/b1 are the
// two 16-byte per-lane chunks of the 16-element bf16 fragments.
__device__ __forceinline__ f32x8 wmma_bf16(u16x8 a0, u16x8 a1,
                                           u16x8 b0, u16x8 b1, f32x8 c) {
  u16x16 av = __builtin_shufflevector(a0, a1, 0,1,2,3,4,5,6,7,8,9,10,11,12,13,14,15);
  u16x16 bv = __builtin_shufflevector(b0, b1, 0,1,2,3,4,5,6,7,8,9,10,11,12,13,14,15);
  return __builtin_amdgcn_wmma_f32_16x16x32_bf16(
      false, __builtin_bit_cast(bf16x16, av),
      false, __builtin_bit_cast(bf16x16, bv),
      (short)0, c, false, false);
}

// CDNA5 async global->LDS DMA (ASYNCcnt-tracked), 16B granule.
__device__ __forceinline__ void async_ld_b128(void* lds_dst, const void* gsrc) {
  unsigned ldsAddr = (unsigned)(uintptr_t)lds_dst;   // low 32 bits = LDS offset
  asm volatile("global_load_async_to_lds_b128 %0, %1, off"
               :: "v"(ldsAddr), "v"(gsrc)
               : "memory");
}
__device__ __forceinline__ void wait_asynccnt0() {
  asm volatile("s_wait_asynccnt 0x0" ::: "memory");
}

// ---------------------------------------------------------------------------
// Problem constants
// ---------------------------------------------------------------------------
constexpr int Bc  = 4;
constexpr int Tc  = 1024;
constexpr int TMc = 1500;
constexpr int TMP = 1504;      // padded Tk for VT rows (16B-aligned b128 loads)
constexpr int DMc = 1024;
constexpr int DFc = 4096;
constexpr int Hc  = 16;
constexpr int DHc = 64;

// ---------------------------------------------------------------------------
// fp32 -> bf16 convert
// ---------------------------------------------------------------------------
__global__ void cvt_kernel(const float* __restrict__ in, u16* __restrict__ out, int n) {
  int i = blockIdx.x * blockDim.x + threadIdx.x;
  int stride = gridDim.x * blockDim.x;
  for (; i < n; i += stride) out[i] = f2bf(in[i]);
}

// ---------------------------------------------------------------------------
// LayerNorm row kernel: out_bf16 = (x - mean) * rsqrt(var + eps) * g
// ---------------------------------------------------------------------------
__global__ __launch_bounds__(256)
void ln_kernel(const float* __restrict__ X, const float* __restrict__ g,
               u16* __restrict__ O, int D) {
  __shared__ float s1[256];
  __shared__ float s2[256];
  const int row = blockIdx.x, tid = threadIdx.x;
  const float* xr = X + (size_t)row * D;
  float a = 0.f, b = 0.f;
  for (int i = tid; i < D; i += 256) { float v = xr[i]; a += v; b += v * v; }
  s1[tid] = a; s2[tid] = b;
  __syncthreads();
  for (int s = 128; s > 0; s >>= 1) {
    if (tid < s) { s1[tid] += s1[tid + s]; s2[tid] += s2[tid + s]; }
    __syncthreads();
  }
  float mean = s1[0] / (float)D;
  float var  = s2[0] / (float)D - mean * mean;
  float rs   = rsqrtf(var + 1e-5f);
  for (int i = tid; i < D; i += 256)
    O[(size_t)row * D + i] = f2bf((xr[i] - mean) * rs * g[i]);
}

// ---------------------------------------------------------------------------
// WMMA GEMM: C(MxN,f32) = A(MxK,bf16 row-major) * B(KxN,bf16 row-major)
// Block tile 128x128, K-step 32, 8 waves (4M x 2N), wave tile 32x64.
// A tile staged via global_load_async_to_lds_b128 (ASYNCcnt), B tile staged
// transposed to [n][k] via regular loads (needs transposition).
// EPI: 0 = f32 store, 1 = f32 residual add, 2 = bf16 store, 3 = gelu->bf16
// ---------------------------------------------------------------------------
#define LDP 40   // padded LDS K-stride (elements); 40*2=80B keeps 16B alignment

template<int EPI>
__global__ __launch_bounds__(256)
void gemm_kernel(const u16* __restrict__ A, const u16* __restrict__ B,
                 const float* __restrict__ Res, float* __restrict__ Of,
                 u16* __restrict__ Ob, int M, int N, int K) {
  __shared__ u16 As[128 * LDP];
  __shared__ u16 Bs[128 * LDP];
  const int tid  = threadIdx.x;
  const int lane = tid & 31;
  const int w    = tid >> 5;
  const int l15  = lane & 15;
  const int hi   = (lane & 16) ? 1 : 0;
  const int rowBase = blockIdx.y * 128;
  const int colBase = blockIdx.x * 128;
  const int wm = (w & 3) * 32;
  const int wn = (w >> 2) * 64;

  f32x8 acc[2][4];
#pragma unroll
  for (int i = 0; i < 2; ++i)
#pragma unroll
    for (int j = 0; j < 4; ++j) acc[i][j] = zero8();

  for (int k0 = 0; k0 < K; k0 += 32) {
    // Stage A tile (128 x 32) row-major: async DMA straight into LDS
    for (int i = tid; i < 512; i += 256) {
      int r = i >> 2, cv = (i & 3) << 3;
      int gr = rowBase + r;
      u16* dst = As + r * LDP + cv;
      if (gr < M) {
        async_ld_b128(dst, A + (size_t)gr * K + k0 + cv);
      } else {
        u16x8 z = {0, 0, 0, 0, 0, 0, 0, 0};
        *(u16x8*)dst = z;
      }
    }
    // Stage B tile transposed to [n][k]
    for (int i = tid; i < 512; i += 256) {
      int kr = i >> 4;
      int nv = (i & 15) << 3;
      int gc = colBase + nv;
      u16x8 v = {0, 0, 0, 0, 0, 0, 0, 0};
      if (gc + 7 < N) v = *(const u16x8*)(B + (size_t)(k0 + kr) * N + gc);
#pragma unroll
      for (int j = 0; j < 8; ++j) Bs[(nv + j) * LDP + kr] = v[j];
    }
    // Prefetch next K tiles while we compute
    if (k0 + 32 < K) {
      int gr = rowBase + (tid >> 1);
      if (gr < M) __builtin_prefetch(A + (size_t)gr * K + k0 + 32 + (tid & 1) * 16, 0, 1);
      int gc = colBase + (tid & 127);
      if (gc < N) __builtin_prefetch(B + (size_t)(k0 + 32 + (tid >> 7) * 16) * N + gc, 0, 1);
    }
    wait_asynccnt0();     // drain this wave's async DMA before the barrier
    __syncthreads();

    // Fragments in contiguous k order (k = 16*hi + s), same for A and B
    u16x8 aF[2][2];
    const int ak = hi * 16;
#pragma unroll
    for (int tm = 0; tm < 2; ++tm) {
      const u16* p = As + (wm + tm * 16 + l15) * LDP;
      aF[tm][0] = *(const u16x8*)(p + ak);        // K = ak .. ak+7
      aF[tm][1] = *(const u16x8*)(p + ak + 8);    // K = ak+8 .. ak+15
    }
    u16x8 bF[4][2];
    const int bk = hi * 16;
#pragma unroll
    for (int tn = 0; tn < 4; ++tn) {
      const u16* p = Bs + (wn + tn * 16 + l15) * LDP + bk;
      bF[tn][0] = *(const u16x8*)(p);             // K = bk .. bk+7
      bF[tn][1] = *(const u16x8*)(p + 8);         // K = bk+8 .. bk+15
    }
#pragma unroll
    for (int tm = 0; tm < 2; ++tm)
#pragma unroll
      for (int tn = 0; tn < 4; ++tn)
        acc[tm][tn] = wmma_bf16(aF[tm][0], aF[tm][1], bF[tn][0], bF[tn][1], acc[tm][tn]);
    __syncthreads();
  }

  // Epilogue: C element (m,n): lane = n%16 (+16 for m%16>=8), vgpr r = m%8
#pragma unroll
  for (int tm = 0; tm < 2; ++tm) {
    int gmBase = rowBase + wm + tm * 16 + hi * 8;
#pragma unroll
    for (int tn = 0; tn < 4; ++tn) {
      int gn = colBase + wn + tn * 16 + l15;
      if (gn >= N) continue;
#pragma unroll
      for (int r = 0; r < 8; ++r) {
        int gm = gmBase + r;
        if (gm >= M) continue;
        size_t idx = (size_t)gm * N + gn;
        float v = acc[tm][tn][r];
        if (EPI == 0) {
          Of[idx] = v;
        } else if (EPI == 1) {
          Of[idx] = Res[idx] + v;
        } else if (EPI == 2) {
          Ob[idx] = f2bf(v);
        } else {
          float t = 0.7978845608028654f * (v + 0.044715f * v * v * v);
          Ob[idx] = f2bf(0.5f * v * (1.f + tanhf(t)));
        }
      }
    }
  }
}

// ---------------------------------------------------------------------------
// Split qkv: fp32 (B,T,3,H,64) -> q,k bf16 [B,H,T,64], vT bf16 [B,H,64,T],
// plus fp32 k,v outputs in (B,T,H,64).
// ---------------------------------------------------------------------------
__global__ void split_qkv_kernel(const float* __restrict__ qkv,
                                 u16* __restrict__ qh, u16* __restrict__ kh,
                                 u16* __restrict__ vT,
                                 float* __restrict__ outK, float* __restrict__ outV) {
  int idx = blockIdx.x * blockDim.x + threadIdx.x;
  if (idx >= Bc * Tc * Hc * DHc) return;
  int d = idx & 63, h = (idx >> 6) & 15, t = (idx >> 10) & 1023, b = idx >> 20;
  size_t base = ((size_t)(b * Tc + t) * 3) * (Hc * DHc) + h * DHc + d;
  float qv = qkv[base];
  float kv = qkv[base + Hc * DHc];
  float vv = qkv[base + 2 * Hc * DHc];
  size_t hoff = ((size_t)(b * Hc + h) * Tc + t) * DHc + d;
  qh[hoff] = f2bf(qv);
  kh[hoff] = f2bf(kv);
  vT[((size_t)(b * Hc + h) * DHc + d) * Tc + t] = f2bf(vv);
  size_t ko = ((size_t)(b * Tc + t) * Hc + h) * DHc + d;
  outK[ko] = kv;
  outV[ko] = vv;
}

// Reorder bf16 (B,T,H*64) -> (B,H,T,64)
__global__ void reorder_q_kernel(const u16* __restrict__ in, u16* __restrict__ out) {
  int idx = blockIdx.x * blockDim.x + threadIdx.x;
  if (idx >= Bc * Tc * Hc * DHc) return;
  int d = idx & 63, h = (idx >> 6) & 15, t = (idx >> 10) & 1023, b = idx >> 20;
  out[((size_t)(b * Hc + h) * Tc + t) * DHc + d] =
      in[((size_t)(b * Tc + t)) * (Hc * DHc) + h * DHc + d];
}

// Split cross kv: fp32 (B,TM,2,H,64) -> kx bf16 [B,H,TM,64], vxT bf16 [B,H,64,TMP]
__global__ void split_kv_kernel(const float* __restrict__ kvx,
                                u16* __restrict__ kx, u16* __restrict__ vxT) {
  int idx = blockIdx.x * blockDim.x + threadIdx.x;
  if (idx >= Bc * TMc * Hc * DHc) return;
  int d = idx & 63, h = (idx >> 6) & 15;
  int tmp = idx >> 10;
  int tm = tmp % TMc, b = tmp / TMc;
  size_t base = ((size_t)(b * TMc + tm) * 2) * (Hc * DHc) + h * DHc + d;
  kx[((size_t)(b * Hc + h) * TMc + tm) * DHc + d] = f2bf(kvx[base]);
  vxT[((size_t)(b * Hc + h) * DHc + d) * TMP + tm] = f2bf(kvx[base + Hc * DHc]);
}

// ---------------------------------------------------------------------------
// Flash attention, one wave per 16-query tile (4 waves / block).
// Computes S^T = K_tile(16x64) @ Q^T(64x16) so keys land in the VGPR (M) axis:
// row softmax = 8-VGPR in-lane max/sum + one shfl_xor(16).
// Then O(16x64) += P(16x32) @ V(32x64) via LDS-staged P.
// Q,K: [B,H,Tq/Tk,64] bf16.  VT: [B,H,64,vtStride] bf16.  Out: [B,Tq,H*64] bf16.
// ---------------------------------------------------------------------------
template<bool CAUSAL>
__global__ __launch_bounds__(128)
void attn_kernel(const u16* __restrict__ Q, const u16* __restrict__ Kb,
                 const u16* __restrict__ VT, const unsigned char* __restrict__ emask,
                 u16* __restrict__ Out, int Tq, int Tk, int vtStride, float scale) {
  __shared__ u16 Pl[4][16 * LDP];
  __shared__ float Fb[4][16];
  const int lane = threadIdx.x & 31;
  const int w    = threadIdx.x >> 5;
  const int l15  = lane & 15;
  const int hi   = (lane & 16) ? 1 : 0;
  const int b = blockIdx.z, h = blockIdx.y;
  const int q0 = blockIdx.x * 64 + w * 16;

  const size_t bh = (size_t)(b * Hc + h);
  const u16* Qp = Q + bh * (size_t)Tq * DHc;
  const u16* Kp = Kb + bh * (size_t)Tk * DHc;
  const u16* Vp = VT + bh * (size_t)DHc * vtStride;
  const unsigned char* mb = CAUSAL ? nullptr : (emask + (size_t)b * Tk);

  // Q^T B-fragments: chunk c covers d = 32c..32c+31
  u16x8 qF[2][2];
  {
    const u16* p = Qp + (size_t)(q0 + l15) * DHc + hi * 8;
    qF[0][0] = *(const u16x8*)(p);
    qF[0][1] = *(const u16x8*)(p + 16);
    qF[1][0] = *(const u16x8*)(p + 32);
    qF[1][1] = *(const u16x8*)(p + 48);
  }

  f32x8 o[4];
#pragma unroll
  for (int j = 0; j < 4; ++j) o[j] = zero8();
  float m_run = -3.0e38f, l_run = 0.f;

  int kend = CAUSAL ? (q0 + 16) : Tk;
  if (kend > Tk) kend = Tk;
  const int nkb = (kend + 31) >> 5;

  for (int kb = 0; kb < nkb; ++kb) {
    const int kbase = kb << 5;
    // ---- S^T tiles: keys 0..15 (t=0) and 16..31 (t=1) ----
    f32x8 st[2];
#pragma unroll
    for (int t = 0; t < 2; ++t) {
      int krow = kbase + t * 16 + l15;
      if (krow >= Tk) krow = Tk - 1;   // clamped rows get masked below
      const u16* p = Kp + (size_t)krow * DHc;
      u16x8 k00 = *(const u16x8*)(p + hi * 8);
      u16x8 k01 = *(const u16x8*)(p + hi * 8 + 16);
      u16x8 k10 = *(const u16x8*)(p + 32 + hi * 8);
      u16x8 k11 = *(const u16x8*)(p + 32 + hi * 8 + 16);
      f32x8 c = zero8();
      c = wmma_bf16(k00, k01, qF[0][0], qF[0][1], c);
      c = wmma_bf16(k10, k11, qF[1][0], qF[1][1], c);
      st[t] = c;
    }
    // ---- scale + mask + online softmax ----
    const int qq = q0 + l15;
    float sv[2][8];
    float bmax = -3.0e38f;
#pragma unroll
    for (int t = 0; t < 2; ++t)
#pragma unroll
      for (int r = 0; r < 8; ++r) {
        int key = kbase + t * 16 + r + hi * 8;
        float s = st[t][r] * scale;
        if (CAUSAL) {
          if (key > qq) s = -1.0e9f;
        } else {
          if (key >= Tk || mb[key] == 0) s = -1.0e30f;
        }
        sv[t][r] = s;
        bmax = fmaxf(bmax, s);
      }
    bmax = fmaxf(bmax, __shfl_xor(bmax, 16));
    float m_new = fmaxf(m_run, bmax);
    float corr  = __expf(m_run - m_new);
    float psum  = 0.f;
#pragma unroll
    for (int t = 0; t < 2; ++t)
#pragma unroll
      for (int r = 0; r < 8; ++r) {
        float s = sv[t][r];
        float p = (s < -1.0e28f) ? 0.f : __expf(s - m_new);
        psum += p;
        Pl[w][l15 * LDP + t * 16 + r + hi * 8] = f2bf(p);
      }
    psum += __shfl_xor(psum, 16);
    l_run = l_run * corr + psum;
    m_run = m_new;
    // broadcast per-query correction through LDS (query index = VGPR index)
    if (!hi) Fb[w][l15] = corr;
    float fr[8];
#pragma unroll
    for (int r = 0; r < 8; ++r) fr[r] = Fb[w][r + hi * 8];
#pragma unroll
    for (int j = 0; j < 4; ++j)
#pragma unroll
      for (int r = 0; r < 8; ++r) o[j][r] *= fr[r];
    // ---- O += P @ V ----
    const u16* pp = &Pl[w][l15 * LDP];
    u16x8 pf0 = *(const u16x8*)(pp + hi * 16);
    u16x8 pf1 = *(const u16x8*)(pp + hi * 16 + 8);
#pragma unroll
    for (int j = 0; j < 4; ++j) {
      const u16* vp = Vp + (size_t)(j * 16 + l15) * vtStride + kbase + hi * 16;
      u16x8 v0 = *(const u16x8*)(vp);
      u16x8 v1 = *(const u16x8*)(vp + 8);
      o[j] = wmma_bf16(pf0, pf1, v0, v1, o[j]);
    }
  }

  // ---- finalize: O /= l, store bf16 to (B,Tq,H*64) ----
  float linv = 1.f / l_run;
  if (!hi) Fb[w][l15] = linv;
  float fr[8];
#pragma unroll
  for (int r = 0; r < 8; ++r) fr[r] = Fb[w][r + hi * 8];
  const int outStride = Hc * DHc;
#pragma unroll
  for (int j = 0; j < 4; ++j) {
    int d = j * 16 + l15;
#pragma unroll
    for (int r = 0; r < 8; ++r) {
      int q = q0 + r + hi * 8;
      Out[((size_t)b * Tq + q) * outStride + h * DHc + d] = f2bf(o[j][r] * fr[r]);
    }
  }
}

// ---------------------------------------------------------------------------
// Host-side orchestration
// ---------------------------------------------------------------------------
extern "C" void kernel_launch(void* const* d_in, const int* in_sizes, int n_in,
                              void* d_out, int out_size, void* d_ws, size_t ws_size,
                              hipStream_t stream) {
  (void)in_sizes; (void)n_in; (void)out_size; (void)ws_size;

  const float* x      = (const float*)d_in[0];
  const float* enc    = (const float*)d_in[1];
  const unsigned char* emask = (const unsigned char*)d_in[2];
  const float* w_qkv  = (const float*)d_in[3];
  const float* w_o_sa = (const float*)d_in[4];
  const float* w_q    = (const float*)d_in[5];
  const float* w_kv   = (const float*)d_in[6];
  const float* w_o_xa = (const float*)d_in[7];
  const float* w_ff1  = (const float*)d_in[8];
  const float* w_ff2  = (const float*)d_in[9];
  const float* g_sa   = (const float*)d_in[10];
  const float* g_xaq  = (const float*)d_in[11];
  const float* g_xam  = (const float*)d_in[12];
  const float* g_ff   = (const float*)d_in[13];

  float* outX = (float*)d_out;
  float* outK = outX + (size_t)Bc * Tc * Hc * DHc;
  float* outV = outK + (size_t)Bc * Tc * Hc * DHc;

  // ---- workspace bump allocator ----
  char* wp = (char*)d_ws;
  auto alloc = [&](size_t bytes) -> char* {
    char* r = wp;
    wp += (bytes + 255) & ~(size_t)255;
    return r;
  };
  const size_t BT = (size_t)Bc * Tc;           // 4096
  const size_t BM = (size_t)Bc * TMc;          // 6000
  u16* wqkv_b = (u16*)alloc((size_t)DMc * 3 * Hc * DHc * 2);
  u16* wosa_b = (u16*)alloc((size_t)Hc * DHc * DMc * 2);
  u16* wq_b   = (u16*)alloc((size_t)DMc * Hc * DHc * 2);
  u16* wox_b  = (u16*)alloc((size_t)Hc * DHc * DMc * 2);
  u16* wkv_b  = (u16*)alloc((size_t)DMc * 2 * Hc * DHc * 2);
  u16* wff1_b = (u16*)alloc((size_t)DMc * DFc * 2);
  u16* wff2_b = (u16*)alloc((size_t)DFc * DMc * 2);
  u16* lnbuf  = (u16*)alloc(BT * DMc * 2);                 // xn / qn / hn (sequential reuse)
  u16* mn_b   = (u16*)alloc(BM * DMc * 2);
  u16* qh     = (u16*)alloc(BT * DMc * 2);                 // self q, later cross q
  u16* kh     = (u16*)alloc(BT * DMc * 2);
  u16* vT     = (u16*)alloc(BT * DMc * 2);
  u16* actbuf = (u16*)alloc(BT * DMc * 2);                 // sa / qx_rm / xa (sequential reuse)
  u16* kx_b   = (u16*)alloc(BM * Hc * DHc * 2);
  u16* vxT_b  = (u16*)alloc((size_t)Bc * Hc * DHc * TMP * 2);
  u16* h1_b   = (u16*)alloc(BT * DFc * 2);
  float* big  = (float*)alloc(BT * 3 * Hc * DHc * 4);      // qkv f32, later kvx f32
  float* x1   = (float*)alloc(BT * DMc * 4);
  float* x2   = (float*)alloc(BT * DMc * 4);

  const float scale = 0.125f;   // 1/sqrt(64)

  // ---- 0. weight conversions ----
  cvt_kernel<<<2048, 256, 0, stream>>>(w_qkv,  wqkv_b, DMc * 3 * Hc * DHc);
  cvt_kernel<<<2048, 256, 0, stream>>>(w_o_sa, wosa_b, Hc * DHc * DMc);
  cvt_kernel<<<2048, 256, 0, stream>>>(w_q,    wq_b,   DMc * Hc * DHc);
  cvt_kernel<<<2048, 256, 0, stream>>>(w_o_xa, wox_b,  Hc * DHc * DMc);
  cvt_kernel<<<2048, 256, 0, stream>>>(w_kv,   wkv_b,  DMc * 2 * Hc * DHc);
  cvt_kernel<<<2048, 256, 0, stream>>>(w_ff1,  wff1_b, DMc * DFc);
  cvt_kernel<<<2048, 256, 0, stream>>>(w_ff2,  wff2_b, DFc * DMc);

  // ---- 1. self-attention ----
  ln_kernel<<<(int)BT, 256, 0, stream>>>(x, g_sa, lnbuf, DMc);
  gemm_kernel<0><<<dim3(24, 32), 256, 0, stream>>>(lnbuf, wqkv_b, nullptr, big, nullptr,
                                                   (int)BT, 3 * Hc * DHc, DMc);
  {
    int n = Bc * Tc * Hc * DHc;
    split_qkv_kernel<<<(n + 255) / 256, 256, 0, stream>>>(big, qh, kh, vT, outK, outV);
  }
  attn_kernel<true><<<dim3(Tc / 64, Hc, Bc), 128, 0, stream>>>(
      qh, kh, vT, nullptr, actbuf, Tc, Tc, Tc, scale);
  gemm_kernel<1><<<dim3(8, 32), 256, 0, stream>>>(actbuf, wosa_b, x, x1, nullptr,
                                                  (int)BT, DMc, Hc * DHc);

  // ---- 2. cross-attention ----
  ln_kernel<<<(int)BT, 256, 0, stream>>>(x1, g_xaq, lnbuf, DMc);
  ln_kernel<<<(int)BM, 256, 0, stream>>>(enc, g_xam, mn_b, DMc);
  gemm_kernel<2><<<dim3(8, 32), 256, 0, stream>>>(lnbuf, wq_b, nullptr, nullptr, actbuf,
                                                  (int)BT, Hc * DHc, DMc);
  {
    int n = Bc * Tc * Hc * DHc;
    reorder_q_kernel<<<(n + 255) / 256, 256, 0, stream>>>(actbuf, qh);
  }
  gemm_kernel<0><<<dim3(16, 47), 256, 0, stream>>>(mn_b, wkv_b, nullptr, big, nullptr,
                                                   (int)BM, 2 * Hc * DHc, DMc);
  hipMemsetAsync(vxT_b, 0, (size_t)Bc * Hc * DHc * TMP * 2, stream);  // clear pad cols
  {
    int n = Bc * TMc * Hc * DHc;
    split_kv_kernel<<<(n + 255) / 256, 256, 0, stream>>>(big, kx_b, vxT_b);
  }
  attn_kernel<false><<<dim3(Tc / 64, Hc, Bc), 128, 0, stream>>>(
      qh, kx_b, vxT_b, emask, actbuf, Tc, TMc, TMP, scale);
  gemm_kernel<1><<<dim3(8, 32), 256, 0, stream>>>(actbuf, wox_b, x1, x2, nullptr,
                                                  (int)BT, DMc, Hc * DHc);

  // ---- 3. FFN ----
  ln_kernel<<<(int)BT, 256, 0, stream>>>(x2, g_ff, lnbuf, DMc);
  gemm_kernel<3><<<dim3(32, 32), 256, 0, stream>>>(lnbuf, wff1_b, nullptr, nullptr, h1_b,
                                                   (int)BT, DFc, DMc);
  gemm_kernel<1><<<dim3(8, 32), 256, 0, stream>>>(h1_b, wff2_b, x2, outX, nullptr,
                                                  (int)BT, DMc, DFc);
}
